// mRNN_10514079940850
// MI455X (gfx1250) — matrix operand order repaired and checked by reading, not compiled, not run on hardware
//
#include <hip/hip_runtime.h>
#include <hip/hip_bf16.h>

#define N_TOT   700
#define RSTR    768    // r_lds row stride (padded N for 48 uniform tiles)
#define NTILES  48     // 768 / 16 output-neuron tiles -> 3 per wave, uniform
#define KTILES  22     // 704 / 32 reduction tiles (k < 704 only is ever read)
#define T_STEPS 1000
#define INP_DIM 100

typedef __attribute__((ext_vector_type(16))) __bf16 v16bf;
typedef __attribute__((ext_vector_type(8)))  float  v8f;

// Load a 16 x bf16 WMMA fragment as two b128s.  SECOND = element offset of the
// upper 8 halves: 8 for B-fragments (contiguous 16 K values per lane),
// 8 for A-fragments too (same contiguous order: K[kb..kb+15] per lane, as B).
template <int SECOND>
__device__ __forceinline__ v16bf load_frag(const __bf16* p) {
    union { uint4 u[2]; v16bf v; } x;
    x.u[0] = *(const uint4*)(p);
    x.u[1] = *(const uint4*)(p + SECOND);
    return x.v;
}

// One-time: W_eff = relu(W_rec)*mask*sign + fixed, converted to bf16 and
// scattered into per-lane WMMA B-fragment layout:
//   frag (j,kt): lane L -> column n = j*16 + L%16,
//                K = kt*32 + (L<16 ? 0 : 16) + e, e=0..15  (contiguous per lane)
__global__ void pack_weff(const float* __restrict__ Wr, const float* __restrict__ Wm,
                          const float* __restrict__ Ws, const float* __restrict__ Wf,
                          __bf16* __restrict__ Wp)
{
    const int idx = blockIdx.x * blockDim.x + threadIdx.x;   // (j*KTILES+kt)*32 + lane
    if (idx >= NTILES * KTILES * 32) return;
    const int lane = idx & 31;
    const int frag = idx >> 5;
    const int kt   = frag % KTILES;
    const int j    = frag / KTILES;
    const int n    = j * 16 + (lane & 15);
    const int kb   = kt * 32 + ((lane < 16) ? 0 : 16);

    union { __bf16 e[16]; uint4 u[2]; } x;
    #pragma unroll
    for (int q = 0; q < 16; ++q) {
        const int k = kb + q;
        float v = 0.0f;
        if (n < N_TOT && k < N_TOT) {
            const size_t o = (size_t)n * N_TOT + k;
            v = fmaxf(Wr[o], 0.0f) * Wm[o] * Ws[o] + Wf[o];
        }
        x.e[q] = (__bf16)v;
    }
    uint4* dst = (uint4*)(Wp + (size_t)idx * 16);
    dst[0] = x.u[0];
    dst[1] = x.u[1];
}

// Persistent RNN: one block per 16 batch rows, h kept in registers for all T.
// 16 waves; wave w owns N-tiles {w, w+16, w+32} (uniform -> everything unrolls).
__global__ __launch_bounds__(512) void rnn_persistent(
    const float* __restrict__ inp,   const float* __restrict__ noise,
    const float* __restrict__ tonic, const float* __restrict__ out_w,
    const __bf16* __restrict__ Wp,   float* __restrict__ out, int T)
{
    __shared__ __align__(16) __bf16 r_lds[16 * RSTR];   // relu(h_t), bf16
    __shared__ float ow[INP_DIM];

    const int tid  = threadIdx.x;
    const int wave = tid >> 5;
    const int lane = tid & 31;
    const int b0   = blockIdx.x * 16;

    if (tid < INP_DIM) ow[tid] = out_w[tid];

    const int nloc  = lane & 15;              // col within N-tile / A-matrix row M
    const int mrow  = (lane < 16) ? 0 : 8;    // C/D layout: M = g (+8 for hi lanes)
    const int khalf = (lane < 16) ? 0 : 16;   // A layout K split

    int   jj[3];
    float tn[3];
    #pragma unroll
    for (int i = 0; i < 3; ++i) {
        const int j = wave + 16 * i;
        jj[i] = j;
        const int n = j * 16 + nloc;
        tn[i] = (n < N_TOT) ? tonic[n] : 0.0f;
    }

    float h[3][8];
    #pragma unroll
    for (int i = 0; i < 3; ++i)
        #pragma unroll
        for (int g = 0; g < 8; ++g) h[i][g] = 0.0f;

    for (int t = 0; t < T; ++t) {
        // r = relu(h_t) -> LDS (bf16); each (m,n) owned by exactly one lane/vgpr
        #pragma unroll
        for (int i = 0; i < 3; ++i) {
            const int n = jj[i] * 16 + nloc;
            #pragma unroll
            for (int g = 0; g < 8; ++g)
                r_lds[(mrow + g) * RSTR + n] = (__bf16)fmaxf(h[i][g], 0.0f);
        }
        __syncthreads();

        // out[:, t-1] = relu(h_t)[ALM] @ out_w : wave w reduces batch row w
        if (t > 0) {
            float s = 0.0f;
            #pragma unroll
            for (int c = 0; c < 4; ++c) {
                const int col = lane + 32 * c;
                if (col < INP_DIM)
                    s += (float)r_lds[wave * RSTR + 500 + col] * ow[col];
            }
            #pragma unroll
            for (int off = 16; off > 0; off >>= 1) s += __shfl_xor(s, off, 32);
            if (lane == 0)
                __builtin_nontemporal_store(s, &out[(size_t)(b0 + wave) * T + (t - 1)]);
        }

        // Launder a zero offset (not the pointer!) so the 132 B-fragment loads
        // cannot be LICM-hoisted out of the time loop, while keeping the
        // pointer's global address-space provenance -> global_load_b128,
        // LOADcnt-only (a laundered pointer degrades to flat_load, which also
        // ticks DScnt and serializes against the LDS A-fragment stream).
        size_t w_off = 0;
        asm volatile("" : "+s"(w_off));
        const __bf16* Wpt = Wp + w_off;

        // drive = relu(h_t) @ W_eff^T : bf16 WMMA, fp32 accumulate
        v8f acc[3] = {};
        #pragma unroll
        for (int kt = 0; kt < KTILES; ++kt) {
            v16bf a = load_frag<8>(&r_lds[nloc * RSTR + kt * 32 + khalf]);
            #pragma unroll
            for (int i = 0; i < 3; ++i) {
                const size_t fb = ((size_t)(jj[i] * KTILES + kt) * 32 + lane) * 16;
                v16bf b = load_frag<8>(Wpt + fb);
                acc[i] = __builtin_amdgcn_wmma_f32_16x16x32_bf16(
                    false, a, false, b, (short)0, acc[i], false, false);
            }
        }
        __syncthreads();   // all r_lds reads done before next step's overwrite

        // h_{t+1} = h + 0.1*(-h + drive + tonic + ext) + 0.01*noise
        #pragma unroll
        for (int i = 0; i < 3; ++i) {
            const int n = jj[i] * 16 + nloc;
            #pragma unroll
            for (int g = 0; g < 8; ++g) {
                const int b = b0 + mrow + g;
                float ns = 0.0f, ext = 0.0f;
                if (n < N_TOT)
                    ns = __builtin_nontemporal_load(&noise[((size_t)b * T + t) * N_TOT + n]);
                if (n >= 600 && n < N_TOT)
                    ext = __builtin_nontemporal_load(&inp[((size_t)b * T + t) * INP_DIM + (n - 600)])
                          + 0.01f * ns;
                const float hv = h[i][g];
                h[i][g] = hv + 0.1f * (-hv + acc[i][g] + tn[i] + ext) + 0.01f * ns;
            }
        }
    }

    // epilogue: out[:, T-1] from relu(h_T)
    #pragma unroll
    for (int i = 0; i < 3; ++i) {
        const int n = jj[i] * 16 + nloc;
        #pragma unroll
        for (int g = 0; g < 8; ++g)
            r_lds[(mrow + g) * RSTR + n] = (__bf16)fmaxf(h[i][g], 0.0f);
    }
    __syncthreads();
    {
        float s = 0.0f;
        #pragma unroll
        for (int c = 0; c < 4; ++c) {
            const int col = lane + 32 * c;
            if (col < INP_DIM)
                s += (float)r_lds[wave * RSTR + 500 + col] * ow[col];
        }
        #pragma unroll
        for (int off = 16; off > 0; off >>= 1) s += __shfl_xor(s, off, 32);
        if (lane == 0)
            __builtin_nontemporal_store(s, &out[(size_t)(b0 + wave) * T + (T - 1)]);
    }
}

extern "C" void kernel_launch(void* const* d_in, const int* in_sizes, int n_in,
                              void* d_out, int out_size, void* d_ws, size_t ws_size,
                              hipStream_t stream)
{
    const float* inp   = (const float*)d_in[0];
    const float* noise = (const float*)d_in[1];
    const float* Wr    = (const float*)d_in[2];
    const float* Wm    = (const float*)d_in[3];
    const float* Ws    = (const float*)d_in[4];
    const float* Wf    = (const float*)d_in[5];
    const float* tonic = (const float*)d_in[6];
    const float* outw  = (const float*)d_in[7];
    float*  out = (float*)d_out;
    __bf16* Wp  = (__bf16*)d_ws;    // 48*22*512 bf16 ~= 1.03 MB packed weights

    const int T = T_STEPS;
    const int B = in_sizes[0] / (T * INP_DIM);   // 128

    const int packN = NTILES * KTILES * 32;
    pack_weff<<<(packN + 255) / 256, 256, 0, stream>>>(Wr, Wm, Ws, Wf, Wp);
    rnn_persistent<<<B / 16, 512, 0, stream>>>(inp, noise, tonic, outw, Wp, out, T);
}
